// ChannelAttention_88948772700351
// MI455X (gfx1250) — matrix operand order repaired and checked by reading, not compiled, not run on hardware
//
#include <hip/hip_runtime.h>

// ---------------------------------------------------------------------------
// ChannelAttention on MI455X (gfx1250, wave32, WMMA bf16 16x16x32)
//   x: [8, 128, 128, 256] f32 -> A = [B, N=16384, C=256]
//   att = A^T A (per batch), row softmax, out = x + beta * A att_sm^T
// Traffic-bound problem (~2.5 GB moved @ 23.3 TB/s); math on bf16 WMMA.
// ---------------------------------------------------------------------------

typedef __attribute__((ext_vector_type(16))) __bf16 v16bf;
typedef __attribute__((ext_vector_type(8)))  float  v8f;

#define B_DIM   8
#define C_DIM   256
#define N_PIX   16384
#define KSTEP   32
#define LROW    40          // LDS row stride in ushorts (80 B: 16B-aligned, conflict-free)
#define KSPLIT  8
#define NSEG    (N_PIX / KSPLIT)   // 2048 pixels per k-split segment

#if __has_builtin(__builtin_amdgcn_global_load_async_to_lds_b128)
#define HAVE_ASYNC_LDS 1
typedef int v4i __attribute__((vector_size(16)));
typedef __attribute__((address_space(1))) v4i* gptr_v4i;   // global (AS1) int4*
typedef __attribute__((address_space(3))) v4i* lptr_v4i;   // LDS (AS3) int4*
#else
#define HAVE_ASYNC_LDS 0
#endif

union Frag { uint4 q[2]; v16bf v; };

// Hardware bf16 convert (RNE); backend pairs these into v_cvt_pk_bf16_f32.
static __device__ __forceinline__ unsigned int pack_bf16(float a, float b) {
    const unsigned short lo = __builtin_bit_cast(unsigned short, (__bf16)a);
    const unsigned short hi = __builtin_bit_cast(unsigned short, (__bf16)b);
    return (unsigned int)lo | ((unsigned int)hi << 16);
}

// A-operand (16x32 bf16, MxK) for rows [wr, wr+16) from LDS [row][k] layout.
static __device__ __forceinline__ v16bf load_afrag(const unsigned short* s, int wr, int l, int hi) {
    const uint4* p = reinterpret_cast<const uint4*>(s + (wr + l) * LROW);
    Frag f;
    f.q[0] = p[hi];         // hi=0: K0..7   hi=1: K8..15
    f.q[1] = p[hi + 2];     // hi=0: K16..23 hi=1: K24..31
    return f.v;
}

// B-operand (32x16 bf16, KxN): lane = column, K contiguous per lane.
static __device__ __forceinline__ v16bf load_bfrag(const unsigned short* s, int col0, int l, int hi) {
    const uint4* p = reinterpret_cast<const uint4*>(s + (col0 + l) * LROW);
    Frag f;
    f.q[0] = p[hi];         // hi=0: K0..7   hi=1: K8..15
    f.q[1] = p[hi + 2];     // hi=0: K16..23 hi=1: K24..31
    return f.v;
}

static __device__ __forceinline__ v8f wmma_bf16(v16bf a, v16bf b, v8f c) {
    return __builtin_amdgcn_wmma_f32_16x16x32_bf16(false, a, false, b, (short)0, c, false, false);
}

// ---------------------------------------------------------------------------
// Kernel 1: partial Gram  attp[s][b] = A_seg^T A_seg  over segment s.
// Grid: KSPLIT * 8 batches * 16 (4x4) tiles of 64x64 = 1024 blocks, 4 waves.
// Software pipelined: global loads for tile k+1 issue before WMMAs of tile k.
// ---------------------------------------------------------------------------
__global__ __launch_bounds__(128) void gram_kernel(const float* __restrict__ x,
                                                   float* __restrict__ attp) {
    __shared__ unsigned short sA[64 * LROW];
    __shared__ unsigned short sB[64 * LROW];

    const int id  = blockIdx.x;
    const int s   = id >> 7;
    const int rem = id & 127;
    const int b   = rem >> 4;
    const int t   = rem & 15;
    const int c0  = (t >> 2) * 64;
    const int d0  = (t & 3)  * 64;
    const float* xb = x + (size_t)b * N_PIX * C_DIM;
    const int kbeg = s * NSEG;
    const int kend = kbeg + NSEG;

    const int tid  = threadIdx.x;
    const int lane = tid & 31;
    const int w    = tid >> 5;
    const int hi   = lane >> 4;
    const int l    = lane & 15;
    const int wr   = w * 16;

    // staging coords: thread owns (kk0,kk0+1) and (kk0+16,kk0+17) at column-block c4
    const int kk0 = (tid >> 4) * 2;
    const int kk1 = kk0 + 16;
    const int c4  = tid & 15;

    float4 a00, a01, b00, b01, a10, a11, b10, b11;

    auto loadT = [&](int k0) {
        const float* r0 = xb + (size_t)(k0 + kk0) * C_DIM;
        a00 = reinterpret_cast<const float4*>(r0 + c0)[c4];
        a01 = reinterpret_cast<const float4*>(r0 + C_DIM + c0)[c4];
        b00 = reinterpret_cast<const float4*>(r0 + d0)[c4];
        b01 = reinterpret_cast<const float4*>(r0 + C_DIM + d0)[c4];
        const float* r2 = xb + (size_t)(k0 + kk1) * C_DIM;
        a10 = reinterpret_cast<const float4*>(r2 + c0)[c4];
        a11 = reinterpret_cast<const float4*>(r2 + C_DIM + c0)[c4];
        b10 = reinterpret_cast<const float4*>(r2 + d0)[c4];
        b11 = reinterpret_cast<const float4*>(r2 + C_DIM + d0)[c4];
    };
    auto storeT = [&]() {
        const int cc = c4 * 4;
        *(unsigned int*)&sA[(cc + 0) * LROW + kk0] = pack_bf16(a00.x, a01.x);
        *(unsigned int*)&sA[(cc + 1) * LROW + kk0] = pack_bf16(a00.y, a01.y);
        *(unsigned int*)&sA[(cc + 2) * LROW + kk0] = pack_bf16(a00.z, a01.z);
        *(unsigned int*)&sA[(cc + 3) * LROW + kk0] = pack_bf16(a00.w, a01.w);
        *(unsigned int*)&sB[(cc + 0) * LROW + kk0] = pack_bf16(b00.x, b01.x);
        *(unsigned int*)&sB[(cc + 1) * LROW + kk0] = pack_bf16(b00.y, b01.y);
        *(unsigned int*)&sB[(cc + 2) * LROW + kk0] = pack_bf16(b00.z, b01.z);
        *(unsigned int*)&sB[(cc + 3) * LROW + kk0] = pack_bf16(b00.w, b01.w);
        *(unsigned int*)&sA[(cc + 0) * LROW + kk1] = pack_bf16(a10.x, a11.x);
        *(unsigned int*)&sA[(cc + 1) * LROW + kk1] = pack_bf16(a10.y, a11.y);
        *(unsigned int*)&sA[(cc + 2) * LROW + kk1] = pack_bf16(a10.z, a11.z);
        *(unsigned int*)&sA[(cc + 3) * LROW + kk1] = pack_bf16(a10.w, a11.w);
        *(unsigned int*)&sB[(cc + 0) * LROW + kk1] = pack_bf16(b10.x, b11.x);
        *(unsigned int*)&sB[(cc + 1) * LROW + kk1] = pack_bf16(b10.y, b11.y);
        *(unsigned int*)&sB[(cc + 2) * LROW + kk1] = pack_bf16(b10.z, b11.z);
        *(unsigned int*)&sB[(cc + 3) * LROW + kk1] = pack_bf16(b10.w, b11.w);
    };

    v8f acc0 = {}, acc1 = {}, acc2 = {}, acc3 = {};

    loadT(kbeg);
    storeT();
    for (int k0 = kbeg; k0 < kend; k0 += KSTEP) {
        __syncthreads();
        const bool more = (k0 + KSTEP) < kend;
        if (more) loadT(k0 + KSTEP);           // hide global latency behind WMMAs

        const v16bf fa = load_afrag(sA, wr, l, hi);
        acc0 = wmma_bf16(fa, load_bfrag(sB,  0, l, hi), acc0);
        acc1 = wmma_bf16(fa, load_bfrag(sB, 16, l, hi), acc1);
        acc2 = wmma_bf16(fa, load_bfrag(sB, 32, l, hi), acc2);
        acc3 = wmma_bf16(fa, load_bfrag(sB, 48, l, hi), acc3);

        __syncthreads();
        if (more) storeT();
    }

    float* ap = attp + (size_t)(s * B_DIM + b) * C_DIM * C_DIM;
#pragma unroll
    for (int r = 0; r < 8; ++r) {
        const int rowc = c0 + wr + hi * 8 + r;
        ap[(size_t)rowc * C_DIM + d0 +  0 + l] = acc0[r];
        ap[(size_t)rowc * C_DIM + d0 + 16 + l] = acc1[r];
        ap[(size_t)rowc * C_DIM + d0 + 32 + l] = acc2[r];
        ap[(size_t)rowc * C_DIM + d0 + 48 + l] = acc3[r];
    }
}

// ---------------------------------------------------------------------------
// Kernel 2: reduce KSPLIT partials + row softmax (last axis) -> bf16 rows.
// One block (256 threads) per row; 2048 rows.
// ---------------------------------------------------------------------------
__global__ __launch_bounds__(256) void softmax_kernel(const float* __restrict__ attp,
                                                      unsigned short* __restrict__ attbf) {
    __shared__ float red[256];
    const int row  = blockIdx.x;        // b*C + c
    const int b    = row >> 8;
    const int crow = row & 255;
    const int tid  = threadIdx.x;

    float v = 0.0f;
#pragma unroll
    for (int s = 0; s < KSPLIT; ++s)
        v += attp[((size_t)(s * B_DIM + b) * C_DIM + crow) * C_DIM + tid];

    red[tid] = v;
    __syncthreads();
    for (int sh = 128; sh > 0; sh >>= 1) {
        if (tid < sh) red[tid] = fmaxf(red[tid], red[tid + sh]);
        __syncthreads();
    }
    const float m = red[0];
    __syncthreads();

    const float e = __expf(v - m);
    red[tid] = e;
    __syncthreads();
    for (int sh = 128; sh > 0; sh >>= 1) {
        if (tid < sh) red[tid] += red[tid + sh];
        __syncthreads();
    }
    const float inv = 1.0f / red[0];
    attbf[(size_t)row * C_DIM + tid] = __builtin_bit_cast(unsigned short, (__bf16)(e * inv));
}

// ---------------------------------------------------------------------------
// Kernel 3: out = x + beta * A @ att_sm^T.  8192 blocks of 64(n)x64(d) tiles.
// A staged fp32->bf16 (pipelined in registers); B (already bf16) staged via
// async global->LDS DMA into a double-buffered tile when available.
// ---------------------------------------------------------------------------
__global__ __launch_bounds__(128) void out_kernel(const float* __restrict__ x,
                                                  const unsigned short* __restrict__ attbf,
                                                  const float* __restrict__ betap,
                                                  float* __restrict__ out) {
    __shared__ unsigned short sA[64 * LROW];
    __shared__ unsigned short sB[2][64 * LROW];

    const int id  = blockIdx.x;
    const int b   = id >> 10;
    const int rem = id & 1023;
    const int n0  = (rem >> 2) * 64;
    const int d0  = (rem & 3)  * 64;

    const int tid  = threadIdx.x;
    const int lane = tid & 31;
    const int w    = tid >> 5;
    const int hi   = lane >> 4;
    const int l    = lane & 15;
    const int wr   = w * 16;

    const float beta = betap[0];
    const float* xb = x + (size_t)b * N_PIX * C_DIM;
    const unsigned short* ab = attbf + (size_t)b * C_DIM * C_DIM;

    // A staging: thread owns rows rA, rA+16, rA+32, rA+48 at float4-column c4
    const int rA = tid >> 3;
    const int c4 = tid & 7;
    float4 va0, va1, va2, va3;
    auto loadA = [&](int kc) {
        va0 = reinterpret_cast<const float4*>(xb + (size_t)(n0 + rA     ) * C_DIM + kc)[c4];
        va1 = reinterpret_cast<const float4*>(xb + (size_t)(n0 + rA + 16) * C_DIM + kc)[c4];
        va2 = reinterpret_cast<const float4*>(xb + (size_t)(n0 + rA + 32) * C_DIM + kc)[c4];
        va3 = reinterpret_cast<const float4*>(xb + (size_t)(n0 + rA + 48) * C_DIM + kc)[c4];
    };
    auto storeA = [&]() {
        const int kkk = c4 * 4;
        uint2 u;
        u.x = pack_bf16(va0.x, va0.y); u.y = pack_bf16(va0.z, va0.w);
        *reinterpret_cast<uint2*>(&sA[(rA     ) * LROW + kkk]) = u;
        u.x = pack_bf16(va1.x, va1.y); u.y = pack_bf16(va1.z, va1.w);
        *reinterpret_cast<uint2*>(&sA[(rA + 16) * LROW + kkk]) = u;
        u.x = pack_bf16(va2.x, va2.y); u.y = pack_bf16(va2.z, va2.w);
        *reinterpret_cast<uint2*>(&sA[(rA + 32) * LROW + kkk]) = u;
        u.x = pack_bf16(va3.x, va3.y); u.y = pack_bf16(va3.z, va3.w);
        *reinterpret_cast<uint2*>(&sA[(rA + 48) * LROW + kkk]) = u;
    };

    // B staging: thread owns row rB, 16-byte chunk pB (bf16 source, no convert)
    const int rB = tid >> 1;
    const int pB = tid & 1;
#if HAVE_ASYNC_LDS
    auto loadB_async = [&](int kc, int buf) {
        __builtin_amdgcn_global_load_async_to_lds_b128(
            (gptr_v4i)(ab + (size_t)(d0 + rB) * C_DIM + kc + pB * 8),
            (lptr_v4i)&sB[buf][rB * LROW + pB * 8],
            0, 0);
    };
#else
    uint4 vb;
    auto loadB  = [&](int kc)  { vb = *reinterpret_cast<const uint4*>(ab + (size_t)(d0 + rB) * C_DIM + kc + pB * 8); };
    auto storeB = [&](int buf) { *reinterpret_cast<uint4*>(&sB[buf][rB * LROW + pB * 8]) = vb; };
#endif

    v8f acc0 = {}, acc1 = {}, acc2 = {}, acc3 = {};

    loadA(0);
    storeA();
#if HAVE_ASYNC_LDS
    loadB_async(0, 0);
    asm volatile("s_wait_asynccnt 0x0" ::: "memory");
#else
    loadB(0);
    storeB(0);
#endif

    for (int i = 0; i < C_DIM / KSTEP; ++i) {
        const int kc  = i * KSTEP;
        const int buf = i & 1;
        __syncthreads();
        const bool more = (i + 1) < (C_DIM / KSTEP);
        if (more) {
            loadA(kc + KSTEP);
#if HAVE_ASYNC_LDS
            loadB_async(kc + KSTEP, buf ^ 1);   // DMA into the other buffer
#else
            loadB(kc + KSTEP);
#endif
        }

        const v16bf fa = load_afrag(sA, wr, l, hi);
        acc0 = wmma_bf16(fa, load_bfrag(sB[buf],  0, l, hi), acc0);
        acc1 = wmma_bf16(fa, load_bfrag(sB[buf], 16, l, hi), acc1);
        acc2 = wmma_bf16(fa, load_bfrag(sB[buf], 32, l, hi), acc2);
        acc3 = wmma_bf16(fa, load_bfrag(sB[buf], 48, l, hi), acc3);

        __syncthreads();
        if (more) {
            storeA();
#if !HAVE_ASYNC_LDS
            storeB(buf ^ 1);
#endif
        }
#if HAVE_ASYNC_LDS
        asm volatile("s_wait_asynccnt 0x0" ::: "memory");
#endif
    }

    float* ob = out + (size_t)b * N_PIX * C_DIM;
#pragma unroll
    for (int r = 0; r < 8; ++r) {
        const size_t rown = (size_t)(n0 + wr + hi * 8 + r) * C_DIM;
        ob[rown + d0 +  0 + l] = xb[rown + d0 +  0 + l] + beta * acc0[r];
        ob[rown + d0 + 16 + l] = xb[rown + d0 + 16 + l] + beta * acc1[r];
        ob[rown + d0 + 32 + l] = xb[rown + d0 + 32 + l] + beta * acc2[r];
        ob[rown + d0 + 48 + l] = xb[rown + d0 + 48 + l] + beta * acc3[r];
    }
}

// ---------------------------------------------------------------------------
extern "C" void kernel_launch(void* const* d_in, const int* in_sizes, int n_in,
                              void* d_out, int out_size, void* d_ws, size_t ws_size,
                              hipStream_t stream) {
    (void)in_sizes; (void)n_in; (void)out_size; (void)ws_size;

    const float* x     = (const float*)d_in[0];
    const float* betap = (const float*)d_in[1];
    float* out         = (float*)d_out;

    float* attp = (float*)d_ws;   // KSPLIT * 8 * 256 * 256 f32 = 16 MB partial Gram tiles
    unsigned short* attbf = (unsigned short*)((char*)d_ws +
                             (size_t)KSPLIT * B_DIM * C_DIM * C_DIM * sizeof(float)); // 1 MB bf16

    gram_kernel   <<<KSPLIT * B_DIM * 16,              128, 0, stream>>>(x, attp);
    softmax_kernel<<<B_DIM * C_DIM,                    256, 0, stream>>>(attp, attbf);
    out_kernel    <<<B_DIM * (N_PIX/64) * (C_DIM/64),  128, 0, stream>>>(x, attbf, betap, out);
}
